// YoloLayer_36266703847635
// MI455X (gfx1250) — compile-verified, matrix operand order checked by reading, not run on hardware
//
#include <hip/hip_runtime.h>
#include <hip/hip_bf16.h>
#include <cstdint>

// ---------------------------------------------------------------------------
// YOLO post-process for MI455X (gfx1250).
//   x: [16, 255, 128, 128] f32  (3 anchors * 85 ch)  = 267 MB  -> HBM-bound,
//   ~11.5 us floor at 23.3 TB/s. Everything after the streaming pass touches
//   only KBs per image, so phase 1 is the whole roofline story.
//
// Phase A: streaming obj/class reduction, async-to-LDS depth-4 pipeline,
//          emits one sortable u64 key per candidate (score|inv_idx|label).
// Phase B: exact top-100 per image (12-bit radix prefilter + argmax passes),
//          decodes boxes only for the 100 winners.
// Phase C: 100x100 IoU (union "area_i+area_j" outer sum via WMMA f16 tiles),
//          serial greedy NMS, padded outputs.
// ---------------------------------------------------------------------------

#define BATCH   16
#define HH      128
#define WW      128
#define NA      3
#define HWSZ    (HH * WW)        // 16384
#define NCAND   (HWSZ * NA)      // 49152
#define NCH     255
#define KSEL    100
#define KPAD    112              // 7 WMMA tiles of 16
#define POOLCAP 2048
#define NMS_T   0.7f
#define SCORE_T 0.05f

typedef __attribute__((ext_vector_type(16))) _Float16 v16h;
typedef __attribute__((ext_vector_type(8)))  float    v8f;

__device__ __forceinline__ float sigmoidf_(float x) {
    return 1.0f / (1.0f + __expf(-x));
}

// ---- CDNA5 async global->LDS copy (ASYNCcnt-tracked) ----------------------
__device__ __forceinline__ void async_ld_b32(const float* g, float* l) {
#if defined(__HIP_DEVICE_COMPILE__)
#if __has_builtin(__builtin_amdgcn_global_load_async_to_lds_b32)
    // Builtin params are addrspace(1) int* / addrspace(3) int* (per hipcc
    // diagnostic: "__device__ int *"). Cast through integers.
    typedef __attribute__((address_space(1))) int as1_int;
    typedef __attribute__((address_space(3))) int as3_int;
    __builtin_amdgcn_global_load_async_to_lds_b32(
        (as1_int*)(uintptr_t)g,
        (as3_int*)(uint32_t)(uintptr_t)l, 0, 0);
#else
    uint32_t lo = (uint32_t)(uintptr_t)l;  // LDS byte offset (low 32 bits)
    asm volatile("global_load_async_to_lds_b32 %0, %1, off"
                 :: "v"(lo), "v"(g) : "memory");
#endif
#else
    (void)g; (void)l;
#endif
}

__device__ __forceinline__ void wait_async_le(int n) {
#if defined(__HIP_DEVICE_COMPILE__)
#if __has_builtin(__builtin_amdgcn_s_wait_asynccnt)
    switch (n) {
        case 0:  __builtin_amdgcn_s_wait_asynccnt(0); break;
        case 1:  __builtin_amdgcn_s_wait_asynccnt(1); break;
        case 2:  __builtin_amdgcn_s_wait_asynccnt(2); break;
        default: __builtin_amdgcn_s_wait_asynccnt(3); break;
    }
#else
    switch (n) {
        case 0:  asm volatile("s_wait_asynccnt 0x0" ::: "memory"); break;
        case 1:  asm volatile("s_wait_asynccnt 0x1" ::: "memory"); break;
        case 2:  asm volatile("s_wait_asynccnt 0x2" ::: "memory"); break;
        default: asm volatile("s_wait_asynccnt 0x3" ::: "memory"); break;
    }
#endif
#else
    (void)n;
#endif
}

// ---------------------------------------------------------------------------
// Phase A: 256 thr/block, 64 blocks/image. Thread owns one (h,w); loops
// 3 anchors x 81 channels. Per-wave quad-buffered async-to-LDS pipeline.
// Key = [sortable(score):32 | (131071-idx):17 | label:7]
// ---------------------------------------------------------------------------
__global__ __launch_bounds__(256)
void score_filter_kernel(const float* __restrict__ x,
                         unsigned long long* __restrict__ keys) {
    __shared__ float sbuf[8][4][32];
    const int wave = threadIdx.x >> 5, lane = threadIdx.x & 31;
    const int b     = blockIdx.x >> 6;          // 64 blocks per image
    const int pbase = (blockIdx.x & 63) * 256;
    const int p     = pbase + threadIdx.x;      // h*W + w
    const float* base0 = x + (size_t)b * NCH * HWSZ + p;

    const int NSTEP = NA * 81;                  // 243 channel slices
    auto gaddr = [&](int s) -> const float* {
        int a = s / 81, j = s - a * 81;         // j==0 -> obj(ch4), else class
        return base0 + (size_t)(a * 85 + 4 + j) * HWSZ;
    };

    async_ld_b32(gaddr(0), &sbuf[wave][0][lane]);
    async_ld_b32(gaddr(1), &sbuf[wave][1][lane]);
    async_ld_b32(gaddr(2), &sbuf[wave][2][lane]);

    float obj_raw = 0.f, maxv = -1e30f;
    int   maxc = 0;
    for (int s = 0; s < NSTEP; ++s) {
        if (s + 3 < NSTEP) async_ld_b32(gaddr(s + 3), &sbuf[wave][(s + 3) & 3][lane]);
        int rem = NSTEP - 1 - s;                // in-order completion: wait slot s
        wait_async_le(rem < 3 ? rem : 3);
        float v = sbuf[wave][s & 3][lane];
        int j = s % 81;
        if (j == 0) { obj_raw = v; maxv = -1e30f; maxc = 0; }
        else if (v > maxv) { maxv = v; maxc = j - 1; }  // strict > : first argmax
        if (j == 80) {
            int   a     = s / 81;
            float score = sigmoidf_(obj_raw) * sigmoidf_(maxv);
            bool  valid = (obj_raw >= 0.0f) && (score >= SCORE_T); // sig(o)>=.5 <=> o>=0
            float sc    = valid ? score : -1.0f;
            uint32_t bits = __float_as_uint(sc);
            uint32_t srt  = (bits & 0x80000000u) ? ~bits : (bits | 0x80000000u);
            int idx = p * NA + a;               // matches [h,w,A] flatten order
            unsigned long long key =
                ((unsigned long long)srt << 32) |
                ((unsigned long long)(131071u - (unsigned)idx) << 7) |
                (unsigned long long)maxc;
            keys[(size_t)b * NCAND + idx] = key;
        }
    }
}

// ---------------------------------------------------------------------------
// Phase B: one block/image. Radix prefilter on top 12 key bits -> pool
// (exact superset of top-100), then 100 strictly-decreasing argmax passes.
// Decodes boxes only for the 100 winners; pads rows 100..111 with zeros.
// ---------------------------------------------------------------------------
__global__ __launch_bounds__(256)
void topk_decode_kernel(const float* __restrict__ x,
                        const float* __restrict__ anchors,
                        const unsigned long long* __restrict__ keys,
                        float* __restrict__ boxbuf,  float* __restrict__ areabuf,
                        float* __restrict__ scorebuf, int* __restrict__ labbuf) {
    const int b = blockIdx.x, tid = threadIdx.x;
    __shared__ int                hist[4096];
    __shared__ unsigned long long pool[POOLCAP];
    __shared__ unsigned long long red[256];
    __shared__ unsigned long long selk[KSEL];
    __shared__ int shThr, shPoolN, shCnt;

    const unsigned long long* kb = keys + (size_t)b * NCAND;

    for (int i = tid; i < 4096; i += 256) hist[i] = 0;
    if (tid == 0) shCnt = 0;
    __syncthreads();
    for (int i = tid; i < NCAND; i += 256)
        atomicAdd(&hist[(int)(kb[i] >> 52)], 1);
    __syncthreads();
    if (tid == 0) {
        int acc = 0, tb = 0;
        for (int bn = 4095; bn >= 0; --bn) {
            acc += hist[bn];
            if (acc >= KSEL) { tb = bn; break; }
        }
        shThr = tb; shPoolN = acc;
    }
    __syncthreads();
    const bool usePool = (shPoolN <= POOLCAP);
    if (usePool) {
        for (int i = tid; i < NCAND; i += 256) {
            unsigned long long kk = kb[i];
            if ((int)(kk >> 52) >= shThr) pool[atomicAdd(&shCnt, 1)] = kk;
        }
    }
    __syncthreads();
    const int poolN = shCnt;

    unsigned long long prev = ~0ull;
    for (int k = 0; k < KSEL; ++k) {
        unsigned long long best = 0ull;
        if (usePool) {
            for (int i = tid; i < poolN; i += 256) {
                unsigned long long kk = pool[i];
                if (kk < prev && kk > best) best = kk;
            }
        } else {  // exact fallback (degenerate distributions)
            for (int i = tid; i < NCAND; i += 256) {
                unsigned long long kk = kb[i];
                if (kk < prev && kk > best) best = kk;
            }
        }
        red[tid] = best;
        __syncthreads();
        for (int off = 128; off > 0; off >>= 1) {
            if (tid < off && red[tid + off] > red[tid]) red[tid] = red[tid + off];
            __syncthreads();
        }
        prev = red[0];
        if (tid == 0) selk[k] = prev;
        __syncthreads();
    }

    if (tid < KPAD) {
        float bx[4] = {0.f, 0.f, 0.f, 0.f};
        float area  = 0.f;
        if (tid < KSEL) {
            unsigned long long key = selk[tid];
            uint32_t srt  = (uint32_t)(key >> 32);
            uint32_t bits = (srt & 0x80000000u) ? (srt ^ 0x80000000u) : ~srt;
            float score = __uint_as_float(bits);
            int idx = 131071 - (int)((key >> 7) & 0x1FFFFu);
            int lab = (int)(key & 0x7Fu);
            int p = idx / NA, a = idx - p * NA;
            int gh = p >> 7, gw = p & 127;
            const float* bp = x + (size_t)(b * NCH + a * 85) * HWSZ + p;
            float tx = bp[0], ty = bp[HWSZ], tw = bp[2 * HWSZ], th = bp[3 * HWSZ];
            float cx = (sigmoidf_(tx) + (float)gw) * (1.0f / WW);   // SCALE_XY==1
            float cy = (sigmoidf_(ty) + (float)gh) * (1.0f / HH);
            float bw = fminf(fmaxf(__expf(tw) * anchors[a * 2 + 0], 0.f), 2.f);
            float bh = fminf(fmaxf(__expf(th) * anchors[a * 2 + 1], 0.f), 2.f);
            float x1 = cx - 0.5f * bw, y1 = cy - 0.5f * bh;
            bx[0] = fminf(fmaxf(x1, 0.f), 1.f);
            bx[1] = fminf(fmaxf(y1, 0.f), 1.f);
            bx[2] = fminf(fmaxf(x1 + bw, 0.f), 1.f);
            bx[3] = fminf(fmaxf(y1 + bh, 0.f), 1.f);
            area  = fmaxf(bx[2] - bx[0], 0.f) * fmaxf(bx[3] - bx[1], 0.f);
            scorebuf[b * KSEL + tid] = score;
            labbuf  [b * KSEL + tid] = lab;
        }
        float* bb = boxbuf + ((size_t)b * KPAD + tid) * 4;
        bb[0] = bx[0]; bb[1] = bx[1]; bb[2] = bx[2]; bb[3] = bx[3];
        areabuf[b * KPAD + tid] = area;
    }
}

// ---------------------------------------------------------------------------
// Phase C: one block/image, 128 thr = 4 waves. 7x7 tiles of 16x16:
// union denominator (area_i + area_j) via v_wmma_f32_16x16x32_f16 rank-2
// outer product, intersection in VALU, then serial greedy NMS.
// ---------------------------------------------------------------------------
__global__ __launch_bounds__(128)
void nms_out_kernel(const float* __restrict__ boxbuf,
                    const float* __restrict__ areabuf,
                    const float* __restrict__ scorebuf,
                    const int*   __restrict__ labbuf,
                    float* __restrict__ out) {
    const int b = blockIdx.x, tid = threadIdx.x;
    const int wave = tid >> 5, lane = tid & 31;
    __shared__ float sB[KPAD][4];
    __shared__ float sA[KPAD];
    __shared__ float sIou[KPAD][KPAD];
    __shared__ int   sKeep[KPAD];
    __shared__ float sScore[KPAD];

    if (tid < KPAD) {
        const float* bb = boxbuf + ((size_t)b * KPAD + tid) * 4;
        sB[tid][0] = bb[0]; sB[tid][1] = bb[1];
        sB[tid][2] = bb[2]; sB[tid][3] = bb[3];
        sA[tid] = areabuf[b * KPAD + tid];
        float sc = (tid < KSEL) ? scorebuf[b * KSEL + tid] : 0.0f;
        sScore[tid] = sc;
        sKeep[tid]  = (tid < KSEL) && (sc > 0.0f);   // cand = top_s > 0
    }
    __syncthreads();

    for (int t = wave; t < 49; t += 4) {             // uniform per wave: EXEC all-1
        int ti = t / 7, tj = t - ti * 7;
        v16h av = {}; v16h bv = {};
        if (lane < 16) {                              // A[m][0]=area_m, A[m][1]=1
            av[0] = (_Float16)sA[ti * 16 + lane];
            av[1] = (_Float16)1.0f;
            bv[0] = (_Float16)1.0f;                   // B[0][n]=1, B[1][n]=area_n
            bv[1] = (_Float16)sA[tj * 16 + lane];
        }
        v8f c = {};
        c = __builtin_amdgcn_wmma_f32_16x16x32_f16(
                false, av, false, bv, (short)0, c, false, false);
#pragma unroll
        for (int r = 0; r < 8; ++r) {                 // D layout: ISA 7.12.2
            int m = ti * 16 + ((lane < 16) ? r : r + 8);
            int n = tj * 16 + (lane & 15);
            float ltx = fmaxf(sB[m][0], sB[n][0]);
            float lty = fmaxf(sB[m][1], sB[n][1]);
            float rbx = fminf(sB[m][2], sB[n][2]);
            float rby = fminf(sB[m][3], sB[n][3]);
            float inter = fmaxf(rbx - ltx, 0.f) * fmaxf(rby - lty, 0.f);
            float uni   = c[r] - inter;               // area_m + area_n - inter
            sIou[m][n]  = inter / fmaxf(uni, 1e-12f);
        }
    }
    __syncthreads();

    for (int i = 0; i < KSEL; ++i) {                  // greedy suppression
        if (tid > i && tid < KSEL && sKeep[i] && sIou[i][tid] > NMS_T)
            sKeep[tid] = 0;
        __syncthreads();
    }

    if (tid < KSEL) {
        int kp = sKeep[tid];
        float* ob = out + ((size_t)b * KSEL + tid) * 4;
        ob[0] = kp ? sB[tid][0] : 0.f;
        ob[1] = kp ? sB[tid][1] : 0.f;
        ob[2] = kp ? sB[tid][2] : 0.f;
        ob[3] = kp ? sB[tid][3] : 0.f;
        out[BATCH * KSEL * 4 + b * KSEL + tid] = kp ? sScore[tid] : 0.f;
        out[BATCH * KSEL * 5 + b * KSEL + tid] = kp ? (float)labbuf[b * KSEL + tid] : -1.0f;
        out[BATCH * KSEL * 6 + b * KSEL + tid] = kp ? 1.0f : 0.f;
    }
}

extern "C" void kernel_launch(void* const* d_in, const int* in_sizes, int n_in,
                              void* d_out, int out_size, void* d_ws, size_t ws_size,
                              hipStream_t stream) {
    const float* x       = (const float*)d_in[0];   // [16,255,128,128]
    const float* anchors = (const float*)d_in[1];   // [3,2]

    char*  ws   = (char*)d_ws;
    size_t off  = 0;
    unsigned long long* keys = (unsigned long long*)(ws + off);
    off += (size_t)BATCH * NCAND * sizeof(unsigned long long);     // 6.29 MB
    float* boxbuf   = (float*)(ws + off); off += (size_t)BATCH * KPAD * 4 * sizeof(float);
    float* areabuf  = (float*)(ws + off); off += (size_t)BATCH * KPAD * sizeof(float);
    float* scorebuf = (float*)(ws + off); off += (size_t)BATCH * KSEL * sizeof(float);
    int*   labbuf   = (int*)  (ws + off); off += (size_t)BATCH * KSEL * sizeof(int);

    score_filter_kernel<<<BATCH * (HWSZ / 256), 256, 0, stream>>>(x, keys);
    topk_decode_kernel <<<BATCH, 256, 0, stream>>>(x, anchors, keys,
                                                   boxbuf, areabuf, scorebuf, labbuf);
    nms_out_kernel     <<<BATCH, 128, 0, stream>>>(boxbuf, areabuf, scorebuf, labbuf,
                                                   (float*)d_out);
}
